// GraphEncoder_29437705846951
// MI455X (gfx1250) — matrix operand order repaired and checked by reading, not compiled, not run on hardware
//
#include <hip/hip_runtime.h>
#include <math.h>

// ---------------- problem constants (match reference) ----------------
constexpr int N_NODES = 20000;
constexpr int N_EDGES = 100000;
constexpr int ATOM_F  = 40;
constexpr int BOND_F  = 10;
constexpr int H       = 32;     // node hidden
constexpr int EH      = 128;    // edge hidden
constexpr int HH      = H * H;  // 1024
constexpr float EPS_   = 1e-5f;
constexpr float NSLOPE = 0.8f;

// ---------------- WMMA types & helpers (gfx1250, wave32) ----------------
typedef __bf16 bf16;
typedef bf16  v16bf __attribute__((ext_vector_type(16)));   // 32 B
typedef bf16  v8bf  __attribute__((ext_vector_type(8)));    // 16 B
typedef float v8f   __attribute__((ext_vector_type(8)));

__device__ inline v8f wmma_bf16(v16bf a, v16bf b, v8f c) {
  // D = A(16x32 bf16) x B(32x16 bf16) + C(16x16 f32)
  return __builtin_amdgcn_wmma_f32_16x16x32_bf16(false, a, false, b, (short)0, c, false, false);
}

// A fragment (16x32 bf16), ISA 7.12.2: lane&15 = M row, hk=(lane>>4)*8.
// Per lane: elements 0..7  = row[k0 + hk + 0..7]
//           elements 8..15 = row[k0 + 16 + hk + 0..7]
// -> two contiguous 16-byte vector loads.
__device__ inline v16bf load_a_row(const bf16* rowptr, int k0, int lane) {
  const int hk = (lane >> 4) << 3;
  v8bf lo = *(const v8bf*)(rowptr + k0 + hk);
  v8bf hi = *(const v8bf*)(rowptr + k0 + 16 + hk);
  return __builtin_shufflevector(lo, hi, 0, 1, 2, 3, 4, 5, 6, 7,
                                          8, 9, 10, 11, 12, 13, 14, 15);
}

// B fragment from fragment-major packed weights:
// pk[((kt*ntiles_n + nt)*32 + lane)*16 + idx], idx<->k = (lane>>4)*8 + (idx&7) + (idx>>3)*16, n = lane&15
__device__ inline v16bf load_b_pk(const bf16* __restrict__ pk, int kt, int nt,
                                  int ntiles_n, int lane) {
  return *(const v16bf*)(pk + ((((size_t)kt * ntiles_n + nt) * 32 + lane) << 4));
}

__device__ inline float sigmoidf_(float x) { return 1.f / (1.f + expf(-x)); }

// ---------------- small utility kernels ----------------
__global__ void zero_kernel(float* p, long n) {
  long i = (long)blockIdx.x * blockDim.x + threadIdx.x;
  if (i < n) p[i] = 0.f;
}

// Repack a K x Ncols f32 weight matrix into WMMA-B fragment-major bf16 layout.
__global__ void pack_b_kernel(const float* __restrict__ W, int Ncols, long total,
                              bf16* __restrict__ out) {
  long i = (long)blockIdx.x * 256 + threadIdx.x;
  if (i >= total) return;
  const int idx  = (int)(i & 15);
  const int lane = (int)((i >> 4) & 31);
  const long tile = i >> 9;
  const int ntn = Ncols >> 4;
  const int nt = (int)(tile % ntn);
  const int kt = (int)(tile / ntn);
  const int k = kt * 32 + ((lane >> 4) << 3) + (idx & 7) + ((idx >> 3) << 4);
  const int n = nt * 16 + (lane & 15);
  out[i] = (bf16)W[(size_t)k * Ncols + n];
}

// One block per column: batch mean/biased-var -> folded BN affine (scale, shift)
__global__ void col_stats_kernel(const float* __restrict__ x, int rows, int cols,
                                 const float* __restrict__ g, const float* __restrict__ b,
                                 float* __restrict__ scale, float* __restrict__ shift) {
  __shared__ float rs1[256], rs2[256];
  const int c = blockIdx.x;
  float s1 = 0.f, s2 = 0.f;
  for (long r = threadIdx.x; r < rows; r += 256) {
    float v = x[r * cols + c];
    s1 += v; s2 += v * v;
  }
  rs1[threadIdx.x] = s1; rs2[threadIdx.x] = s2;
  __syncthreads();
  for (int s = 128; s > 0; s >>= 1) {
    if ((int)threadIdx.x < s) { rs1[threadIdx.x] += rs1[threadIdx.x + s]; rs2[threadIdx.x] += rs2[threadIdx.x + s]; }
    __syncthreads();
  }
  if (threadIdx.x == 0) {
    float mean = rs1[0] / (float)rows;
    float var  = rs2[0] / (float)rows - mean * mean;
    float sc   = g[c] * rsqrtf(var + EPS_);
    scale[c] = sc;
    shift[c] = b[c] - mean * sc;
  }
}

__global__ void finalize_bn_kernel(const float* __restrict__ s1, const float* __restrict__ s2,
                                   const float* __restrict__ g, const float* __restrict__ b,
                                   float* __restrict__ scale, float* __restrict__ shift,
                                   int n, float count) {
  int i = blockIdx.x * blockDim.x + threadIdx.x;
  if (i < n) {
    float mean = s1[i] / count;
    float var  = s2[i] / count - mean * mean;
    float sc   = g[i] * rsqrtf(var + EPS_);
    scale[i] = sc;
    shift[i] = b[i] - mean * sc;
  }
}

__global__ void deg_kernel(const int* __restrict__ dst, float* __restrict__ deg) {
  long e = (long)blockIdx.x * 256 + threadIdx.x;
  if (e < N_EDGES) unsafeAtomicAdd(&deg[dst[e]], 1.0f);
}

// h = BN(x_node)@node_W + node_b   (block: 8 nodes x 32 outputs)
__global__ __launch_bounds__(256) void node_embed_kernel(
    const float* __restrict__ x_node, const float* __restrict__ scN, const float* __restrict__ shN,
    const float* __restrict__ node_W, const float* __restrict__ node_b, float* __restrict__ h) {
  __shared__ float xs[8][ATOM_F];
  const int t = threadIdx.x;
  const long n0 = (long)blockIdx.x * 8;
  for (int i = t; i < 8 * ATOM_F; i += 256) {
    int nl = i / ATOM_F, k = i % ATOM_F;
    long nn = n0 + nl;
    xs[nl][k] = (nn < N_NODES) ? (x_node[nn * ATOM_F + k] * scN[k] + shN[k]) : 0.f;
  }
  __syncthreads();
  const int nl = t >> 5, j = t & 31;
  const long nn = n0 + nl;
  if (nn < N_NODES) {
    float acc = node_b[j];
#pragma unroll
    for (int k = 0; k < ATOM_F; ++k) acc += xs[nl][k] * node_W[k * H + j];
    h[nn * H + j] = acc;
  }
}

// h_edge(bf16) = BN(x_edge)@edge_W + edge_b   (block: 2 edges x 128 outputs)
__global__ __launch_bounds__(256) void edge_embed_kernel(
    const float* __restrict__ x_edge, const float* __restrict__ scE, const float* __restrict__ shE,
    const float* __restrict__ edge_W, const float* __restrict__ edge_b, bf16* __restrict__ h_edge) {
  __shared__ float xs[2][BOND_F];
  const int t = threadIdx.x;
  const long e0 = (long)blockIdx.x * 2;
  if (t < 2 * BOND_F) {
    int el = t / BOND_F, k = t % BOND_F;
    long e = e0 + el;
    xs[el][k] = (e < N_EDGES) ? (x_edge[e * BOND_F + k] * scE[k] + shE[k]) : 0.f;
  }
  __syncthreads();
  const int el = t >> 7, j = t & 127;
  const long e = e0 + el;
  if (e < N_EDGES) {
    float acc = edge_b[j];
#pragma unroll
    for (int k = 0; k < BOND_F; ++k) acc += xs[el][k] * edge_W[k * EH + j];
    h_edge[e * EH + j] = (bf16)acc;
  }
}

// a_act = LeakyReLU(BN1(u)), vectorized 8 bf16 per thread
__global__ void act_kernel(const bf16* __restrict__ u, const float* __restrict__ sc1,
                           const float* __restrict__ sh1, bf16* __restrict__ a_act) {
  const long ngroups = (long)N_EDGES * EH / 8;
  long i8 = (long)blockIdx.x * 256 + threadIdx.x;
  if (i8 >= ngroups) return;
  const long base = i8 * 8;
  const int k0 = (int)(base & (EH - 1));
  v8bf vin = *(const v8bf*)(u + base);
  v8bf vo;
#pragma unroll
  for (int j = 0; j < 8; ++j) {
    float v = (float)vin[j];
    const int k = k0 + j;
    v = sc1[k] * v + sh1[k];
    v = v > 0.f ? v : NSLOPE * v;
    vo[j] = (bf16)v;
  }
  *(v8bf*)(a_act + base) = vo;
}

// ---------------- WMMA GEMM1: u = h_edge @ W1 + b1 (+ BN1 column stats) ----------------
// block = 8 waves, tile 128(M) x 128(N), K=128 in 4 steps of 32. B from packed w1pk.
__global__ __launch_bounds__(256) void gemm1_kernel(
    const bf16* __restrict__ h_edge, const bf16* __restrict__ w1pk, const float* __restrict__ b1,
    bf16* __restrict__ u, float* __restrict__ s1g, float* __restrict__ s2g) {
  __shared__ float ss1[EH], ss2[EH];
  const int t = threadIdx.x;
  if (t < EH) { ss1[t] = 0.f; ss2[t] = 0.f; }
  __syncthreads();

  const int wave = t >> 5, lane = t & 31;
  const int row0 = blockIdx.x * 128 + wave * 16;
  const int m = lane & 15, hi = lane >> 4;
  const int myrow = row0 + m;
  const int crow  = myrow < N_EDGES ? myrow : N_EDGES - 1;
  const bf16* arow = h_edge + (size_t)crow * EH;
  const bool fulltile = (row0 + 16) <= N_EDGES;

  v16bf afr[4];
#pragma unroll
  for (int kt = 0; kt < 4; ++kt) afr[kt] = load_a_row(arow, kt * 32, lane);

  for (int nt = 0; nt < 8; ++nt) {
    const int n = nt * 16 + m;
    const float bv = b1[n];
    v8f acc;
#pragma unroll
    for (int j = 0; j < 8; ++j) acc[j] = bv;
#pragma unroll
    for (int kt = 0; kt < 4; ++kt)
      acc = wmma_bf16(afr[kt], load_b_pk(w1pk, kt, nt, 8, lane), acc);

    if (fulltile) {                       // branch-free fast path (781/782 blocks)
      float vs = 0.f, vq = 0.f;
#pragma unroll
      for (int r = 0; r < 8; ++r) {
        const int row = row0 + r + (hi << 3);
        const float vv = acc[r];
        u[(size_t)row * EH + n] = (bf16)vv;
        vs += vv; vq += vv * vv;
      }
      atomicAdd(&ss1[n], vs);
      atomicAdd(&ss2[n], vq);
    } else {                              // tail wave: per-row predication
      float vs = 0.f, vq = 0.f;
#pragma unroll
      for (int r = 0; r < 8; ++r) {
        const int row = row0 + r + (hi << 3);
        if (row < N_EDGES) {
          const float vv = acc[r];
          u[(size_t)row * EH + n] = (bf16)vv;
          vs += vv; vq += vv * vv;
        }
      }
      atomicAdd(&ss1[n], vs);
      atomicAdd(&ss2[n], vq);
    }
  }
  __syncthreads();
  if (t < EH) { unsafeAtomicAdd(&s1g[t], ss1[t]); unsafeAtomicAdd(&s2g[t], ss2[t]); }
}

// ---------- WMMA GEMM2 stats pass: column sums of v = a_act@W2 + b2 (v not stored) ----------
// grid: (ceil(E/128), 8 col-chunks of 128). A from a_act, B from packed w2pk.
__global__ __launch_bounds__(256) void gemm2_stats_kernel(
    const bf16* __restrict__ a_act, const bf16* __restrict__ w2pk, const float* __restrict__ b2,
    float* __restrict__ s1g, float* __restrict__ s2g) {
  __shared__ float ss1[128], ss2[128];
  const int t = threadIdx.x;
  const int chunk = blockIdx.y;
  if (t < 128) { ss1[t] = 0.f; ss2[t] = 0.f; }
  __syncthreads();

  const int wave = t >> 5, lane = t & 31;
  const int row0 = blockIdx.x * 128 + wave * 16;
  const int m = lane & 15, hi = lane >> 4;
  const int myrow = row0 + m;
  const int crow  = myrow < N_EDGES ? myrow : N_EDGES - 1;
  const bf16* arow = a_act + (size_t)crow * EH;
  const bool fulltile = (row0 + 16) <= N_EDGES;

  v16bf afr[4];
#pragma unroll
  for (int kt = 0; kt < 4; ++kt) afr[kt] = load_a_row(arow, kt * 32, lane);

  for (int ntl = 0; ntl < 8; ++ntl) {
    const int nt = chunk * 8 + ntl;          // global 16-col tile (0..63)
    const int nl = ntl * 16 + m;             // local col within chunk
    const float bv = b2[(size_t)chunk * 128 + nl];
    v8f acc;
#pragma unroll
    for (int j = 0; j < 8; ++j) acc[j] = bv;
#pragma unroll
    for (int kt = 0; kt < 4; ++kt)
      acc = wmma_bf16(afr[kt], load_b_pk(w2pk, kt, nt, 64, lane), acc);

    float vs = 0.f, vq = 0.f;
    if (fulltile) {
#pragma unroll
      for (int r = 0; r < 8; ++r) { const float vv = acc[r]; vs += vv; vq += vv * vv; }
    } else {
#pragma unroll
      for (int r = 0; r < 8; ++r) {
        const int row = row0 + r + (hi << 3);
        if (row < N_EDGES) { const float vv = acc[r]; vs += vv; vq += vv * vv; }
      }
    }
    atomicAdd(&ss1[nl], vs);
    atomicAdd(&ss2[nl], vq);
  }
  __syncthreads();
  if (t < 128) {
    unsafeAtomicAdd(&s1g[chunk * 128 + t], ss1[t]);
    unsafeAtomicAdd(&s2g[chunk * 128 + t], ss2[t]);
  }
}

// ---------- fused message pass: recompute W_e = BN2(a_act@W2+b2) per tile, ----------
// ---------- msg = h[src] . W_e, scatter-add into m. 64 edges per block. ----------
// Key: each lane's 8 message slots (edge el, out col o) are invariant across the
// nt2 loop and the (el,o)->(lane,r) map is a bijection over the 64x32 buffer, so
// messages accumulate in registers and are written once with plain LDS stores.
__global__ __launch_bounds__(256) void msg_kernel(
    const bf16* __restrict__ a_act, const bf16* __restrict__ w2pk, const float* __restrict__ b2,
    const float* __restrict__ sc2, const float* __restrict__ sh2,
    const float* __restrict__ h, const int* __restrict__ src, const int* __restrict__ dst,
    float* __restrict__ mout) {
  __shared__ float sHs[64 * H];            //  8 KB: gathered h[src]
  __shared__ float sMsg[64 * H];           //  8 KB: per-edge messages (written once)
  __shared__ float sSc2[HH], sSh2[HH], sB2[HH];   // 12 KB
  __shared__ int   sDst[64];
  const int t = threadIdx.x;
  const long e0 = (long)blockIdx.x * 64;

  for (int i = t; i < HH; i += 256) { sSc2[i] = sc2[i]; sSh2[i] = sh2[i]; sB2[i] = b2[i]; }
  if (t < 64) {
    long e = e0 + t;
    if (e < N_EDGES) {
      int s = src[e];
      sDst[t] = dst[e];
#pragma unroll
      for (int k = 0; k < H; ++k) sHs[t * H + k] = h[(size_t)s * H + k];
    } else {
      sDst[t] = -1;
#pragma unroll
      for (int k = 0; k < H; ++k) sHs[t * H + k] = 0.f;
    }
  }
  __syncthreads();

  const int wave = t >> 5, lane = t & 31;
  const int rg = wave & 3;      // 16-edge row group
  const int half = wave >> 2;   // output half: o in [0,16) or [16,32)
  const int m = lane & 15, hi = lane >> 4;
  const long myrow = e0 + rg * 16 + m;
  const long crow  = myrow < N_EDGES ? myrow : N_EDGES - 1;   // hs==0 masks tail
  const bf16* arow = a_act + (size_t)crow * EH;

  v16bf afr[4];
#pragma unroll
  for (int kt = 0; kt < 4; ++kt) afr[kt] = load_a_row(arow, kt * 32, lane);

  const int o = half * 16 + m;            // constant output column per lane
  int elr[8];
#pragma unroll
  for (int r = 0; r < 8; ++r) elr[r] = rg * 16 + r + (hi << 3);  // fixed edge rows

  float msgacc[8];
#pragma unroll
  for (int r = 0; r < 8; ++r) msgacc[r] = 0.f;

  for (int nt2 = 0; nt2 < 32; ++nt2) {
    const int nt = nt2 * 2 + half;        // c = 32*nt2 + 16*half + m
    const int c  = nt * 16 + m;           // i = c>>5 = nt2, o = c&31
    const float bv = sB2[c];
    v8f acc;
#pragma unroll
    for (int j = 0; j < 8; ++j) acc[j] = bv;
#pragma unroll
    for (int kt = 0; kt < 4; ++kt)
      acc = wmma_bf16(afr[kt], load_b_pk(w2pk, kt, nt, 64, lane), acc);
    const float scl = sSc2[c], shf = sSh2[c];
#pragma unroll
    for (int r = 0; r < 8; ++r) {
      const float we = scl * acc[r] + shf;            // W_e[e][nt2][o]
      msgacc[r] += sHs[elr[r] * H + nt2] * we;
    }
  }
#pragma unroll
  for (int r = 0; r < 8; ++r) sMsg[elr[r] * H + o] = msgacc[r];  // exclusive slots
  __syncthreads();

  for (int i = t; i < 64 * H; i += 256) {
    int el = i >> 5, oo = i & 31;
    int d = sDst[el];
    if (d >= 0) unsafeAtomicAdd(&mout[(size_t)d * H + oo], sMsg[i]);
  }
}

// ---------------- GRU cell (per-node thread; weights staged in LDS) ----------------
__global__ __launch_bounds__(256) void gru_kernel(
    const float* __restrict__ mbuf, const float* __restrict__ deg, const float* __restrict__ h_in,
    const float* __restrict__ Wih, const float* __restrict__ Whh,
    const float* __restrict__ bih, const float* __restrict__ bhh,
    float* __restrict__ h_out, float* __restrict__ m_clear) {
  __shared__ float sWi[96 * H], sWh[96 * H], sbi[96], sbh[96];
  const int t = threadIdx.x;
  for (int i = t; i < 96 * H; i += 256) { sWi[i] = Wih[i]; sWh[i] = Whh[i]; }
  if (t < 96) { sbi[t] = bih[t]; sbh[t] = bhh[t]; }
  __syncthreads();

  const long node = (long)blockIdx.x * 256 + t;
  if (node >= N_NODES) return;

  float x[H], hh[H], r[H];
  const float dn = fmaxf(deg[node], 1.0f);
#pragma unroll
  for (int k = 0; k < H; ++k) {
    x[k]  = mbuf[node * H + k] / dn;
    hh[k] = h_in[node * H + k];
  }
#pragma unroll
  for (int j = 0; j < H; ++j) {               // reset gate
    float a = sbi[j] + sbh[j];
#pragma unroll
    for (int k = 0; k < H; ++k) a += x[k] * sWi[j * H + k] + hh[k] * sWh[j * H + k];
    r[j] = sigmoidf_(a);
  }
#pragma unroll
  for (int j = 0; j < H; ++j) {               // update + candidate + combine
    float az = sbi[32 + j] + sbh[32 + j];
    float gin = sbi[64 + j];
    float ghn = sbh[64 + j];
#pragma unroll
    for (int k = 0; k < H; ++k) {
      az  += x[k] * sWi[(32 + j) * H + k] + hh[k] * sWh[(32 + j) * H + k];
      gin += x[k]  * sWi[(64 + j) * H + k];
      ghn += hh[k] * sWh[(64 + j) * H + k];
    }
    const float z  = sigmoidf_(az);
    const float nn = tanhf(gin + r[j] * ghn);
    h_out[node * H + j] = (1.f - z) * nn + z * hh[j];
  }
#pragma unroll
  for (int k = 0; k < H; ++k) m_clear[node * H + k] = 0.f;  // ready for next layer
}

// ---------------- host-side launch ----------------
extern "C" void kernel_launch(void* const* d_in, const int* in_sizes, int n_in,
                              void* d_out, int out_size, void* d_ws, size_t ws_size,
                              hipStream_t stream) {
  const float* x_node  = (const float*)d_in[0];
  const float* x_edge  = (const float*)d_in[1];
  const int*   src     = (const int*)d_in[2];
  const int*   dst     = (const int*)d_in[3];
  const float* bn_n_g  = (const float*)d_in[4];
  const float* bn_n_b  = (const float*)d_in[5];
  const float* node_W  = (const float*)d_in[6];
  const float* node_b  = (const float*)d_in[7];
  const float* bn_e_g  = (const float*)d_in[8];
  const float* bn_e_b  = (const float*)d_in[9];
  const float* edge_W  = (const float*)d_in[10];
  const float* edge_b  = (const float*)d_in[11];
  const float* en_W1   = (const float*)d_in[12];
  const float* en_b1   = (const float*)d_in[13];
  const float* en_bn1g = (const float*)d_in[14];
  const float* en_bn1b = (const float*)d_in[15];
  const float* en_W2   = (const float*)d_in[16];
  const float* en_b2   = (const float*)d_in[17];
  const float* en_bn2g = (const float*)d_in[18];
  const float* en_bn2b = (const float*)d_in[19];
  const float* gru_Wih = (const float*)d_in[20];
  const float* gru_Whh = (const float*)d_in[21];
  const float* gru_bih = (const float*)d_in[22];
  const float* gru_bhh = (const float*)d_in[23];

  // ---- workspace layout ----
  float* ws = (float*)d_ws;
  long cur = 0;
  auto alloc_f = [&](long n) { float* p = ws + cur; cur += n; return p; };

  float* m_buf  = alloc_f((long)N_NODES * H);
  float* deg    = alloc_f(N_NODES);
  float* s1_bn1 = alloc_f(EH);
  float* s2_bn1 = alloc_f(EH);
  float* s1_bn2 = alloc_f(HH);
  float* s2_bn2 = alloc_f(HH);
  const long ZERO_N = cur;                     // everything above is zero-init
  float* h_buf  = alloc_f((long)N_NODES * H);
  float* scN = alloc_f(ATOM_F); float* shN = alloc_f(ATOM_F);
  float* scE = alloc_f(BOND_F); float* shE = alloc_f(BOND_F);
  float* sc1 = alloc_f(EH);     float* sh1 = alloc_f(EH);
  float* sc2 = alloc_f(HH);     float* sh2 = alloc_f(HH);
  cur = (cur + 15) & ~15L;                     // 64B align for wide bf16 loads
  bf16* h_edge = (bf16*)(ws + cur); cur += (long)N_EDGES * EH / 2;
  bf16* u_buf  = (bf16*)(ws + cur); cur += (long)N_EDGES * EH / 2;
  bf16* a_act  = (bf16*)(ws + cur); cur += (long)N_EDGES * EH / 2;
  bf16* w1pk   = (bf16*)(ws + cur); cur += (long)EH * EH / 2;
  bf16* w2pk   = (bf16*)(ws + cur); cur += (long)EH * HH / 2;

  // 1) zero accumulators (m, deg, bn sums) — every call, deterministic
  zero_kernel<<<(int)((ZERO_N + 255) / 256), 256, 0, stream>>>(m_buf, ZERO_N);

  // 2) repack weights to WMMA fragment-major bf16 (L2-resident)
  pack_b_kernel<<<(EH * EH + 255) / 256, 256, 0, stream>>>(en_W1, EH, (long)EH * EH, w1pk);
  pack_b_kernel<<<(EH * HH + 255) / 256, 256, 0, stream>>>(en_W2, HH, (long)EH * HH, w2pk);

  // 3) input batchnorm stats -> folded affine
  col_stats_kernel<<<ATOM_F, 256, 0, stream>>>(x_node, N_NODES, ATOM_F, bn_n_g, bn_n_b, scN, shN);
  col_stats_kernel<<<BOND_F, 256, 0, stream>>>(x_edge, N_EDGES, BOND_F, bn_e_g, bn_e_b, scE, shE);

  // 4) embeddings
  node_embed_kernel<<<(N_NODES + 7) / 8, 256, 0, stream>>>(x_node, scN, shN, node_W, node_b, h_buf);
  edge_embed_kernel<<<(N_EDGES + 1) / 2, 256, 0, stream>>>(x_edge, scE, shE, edge_W, edge_b, h_edge);

  // 5) in-degree
  deg_kernel<<<(N_EDGES + 255) / 256, 256, 0, stream>>>(dst, deg);

  // 6) GEMM1 (WMMA): u = h_edge @ W1 + b1, plus BN1 column stats
  gemm1_kernel<<<(N_EDGES + 127) / 128, 256, 0, stream>>>(h_edge, w1pk, en_b1, u_buf, s1_bn1, s2_bn1);
  finalize_bn_kernel<<<1, 128, 0, stream>>>(s1_bn1, s2_bn1, en_bn1g, en_bn1b, sc1, sh1, EH, (float)N_EDGES);

  // 7) activation pass: a_act = LeakyReLU(BN1(u)) in bf16 (WMMA A-operand form)
  act_kernel<<<(int)(((long)N_EDGES * EH / 8 + 255) / 256), 256, 0, stream>>>(u_buf, sc1, sh1, a_act);

  // 8) GEMM2 stats pass (WMMA): BN2 column stats of a_act@W2 + b2 (no materialization)
  gemm2_stats_kernel<<<dim3((N_EDGES + 127) / 128, 8), 256, 0, stream>>>(
      a_act, w2pk, en_b2, s1_bn2, s2_bn2);
  finalize_bn_kernel<<<4, 256, 0, stream>>>(s1_bn2, s2_bn2, en_bn2g, en_bn2b, sc2, sh2, HH, (float)N_EDGES);

  // 9) 3 message-passing + GRU layers; W_e recomputed in-register each layer
  for (int l = 0; l < 3; ++l) {
    msg_kernel<<<(N_EDGES + 63) / 64, 256, 0, stream>>>(
        a_act, w2pk, en_b2, sc2, sh2, h_buf, src, dst, m_buf);
    float* hout = (l == 2) ? (float*)d_out : h_buf;
    gru_kernel<<<(N_NODES + 255) / 256, 256, 0, stream>>>(
        m_buf, deg, h_buf, gru_Wih, gru_Whh, gru_bih, gru_bhh, hout, m_buf);
  }
}
